// LayerNormLSTM_63891933495933
// MI455X (gfx1250) — matrix operand order repaired and checked by reading, not compiled, not run on hardware
//
#include <hip/hip_runtime.h>
#include <hip/hip_bf16.h>
#include <math.h>
#include <stdint.h>

typedef __attribute__((ext_vector_type(16))) __bf16 v16bf;
typedef __attribute__((ext_vector_type(8)))  __bf16 v8bf;
typedef __attribute__((ext_vector_type(8)))  float  v8f;

#define LDP 40   // padded LDS row stride in halves: 80 B = 16B-aligned rows, 20-bank stride

// ---------------------------------------------------------------- helpers
__device__ __forceinline__ float block_reduce_sum(float v, float* sm) {
    const int tid = threadIdx.x;
    sm[tid] = v;
    __syncthreads();
    for (int s = blockDim.x >> 1; s > 0; s >>= 1) {
        if (tid < s) sm[tid] += sm[tid + s];
        __syncthreads();
    }
    float r = sm[0];
    __syncthreads();
    return r;
}

__device__ __forceinline__ float sigmoidf_(float x) { return 1.0f / (1.0f + __expf(-x)); }

// ---------------------------------------------------------------- f32 -> bf16 conversion
__global__ void f32_to_bf16_kernel(const float* __restrict__ in, __bf16* __restrict__ out, int n) {
    int i = blockIdx.x * blockDim.x + threadIdx.x;
    if (i < n) out[i] = (__bf16)in[i];
}

__global__ void init_hc_kernel(const float* __restrict__ hx, const float* __restrict__ cx,
                               float* __restrict__ h, __bf16* __restrict__ hbf,
                               float* __restrict__ c, int n) {
    int i = blockIdx.x * blockDim.x + threadIdx.x;
    if (i < n) { h[i] = hx[i]; hbf[i] = (__bf16)hx[i]; c[i] = cx[i]; }
}

// ---------------------------------------------------------------- WMMA GEMM (async-to-LDS, double-buffered)
// C[M,2048] = A[M,512](bf16, row addr = (r&63)*sA0 + (r>>6)*sA1) x Bw[2048,512](bf16)^T
// grid.x = M/64 tiles, grid.y = 2048/128 tiles. 256 thr = 8 waves, wave tile 16x64.
__global__ __launch_bounds__(256) void wmma_gemm_kernel(
    const __bf16* __restrict__ A, int sA0, int sA1,
    const __bf16* __restrict__ Bw, float* __restrict__ C)
{
    __shared__ alignas(16) __bf16 lA[2][64 * LDP];
    __shared__ alignas(16) __bf16 lB[2][128 * LDP];

    const int tid = threadIdx.x;
    const int m0  = blockIdx.x * 64;
    const int n0  = blockIdx.y * 128;

    const int w  = tid >> 5;         // wave 0..7
    const int l  = tid & 31;         // lane
    const int wm = w >> 1;           // 0..3 : 16-row strip
    const int wn = w & 1;            // 0..1 : 64-col strip
    const int lm = l & 15;
    const int kh = l >> 4;           // K-half select per WMMA lane layout

    // staging assignments
    const int arow = tid >> 2, acg = tid & 3;          // A: 64 rows x 4 groups of 8 bf16 (16 B)
    const int brow = tid & 127, bseg = tid >> 7;       // B: 128 rows x 2 segs of 16 bf16 (32 B)

    const int gar = m0 + arow;
    const __bf16* Ag = A  + (size_t)(gar & 63) * (size_t)sA0 + (size_t)(gar >> 6) * (size_t)sA1 + acg * 8;
    const __bf16* Bg = Bw + (size_t)(n0 + brow) * 512 + bseg * 16;

    // per-thread LDS byte addresses (low 32 bits of generic address = LDS offset)
    const unsigned ldsA0 = (unsigned)(uintptr_t)(&lA[0][arow * LDP + acg * 8]);
    const unsigned ldsB0 = (unsigned)(uintptr_t)(&lB[0][brow * LDP + bseg * 16]);
    const unsigned bufA = 64 * LDP * 2;    // bytes per A buffer
    const unsigned bufB = 128 * LDP * 2;   // bytes per B buffer

    auto stage = [&](int buf, int k0) {
        const unsigned la = ldsA0 + (unsigned)buf * bufA;
        const unsigned lb = ldsB0 + (unsigned)buf * bufB;
        const uint64_t ga = (uint64_t)(uintptr_t)(Ag + k0);
        const uint64_t gb = (uint64_t)(uintptr_t)(Bg + k0);
        asm volatile("global_load_async_to_lds_b128 %0, %1, off"
                     :: "v"(la), "v"(ga) : "memory");
        asm volatile("global_load_async_to_lds_b128 %0, %1, off"
                     :: "v"(lb), "v"(gb) : "memory");
        asm volatile("global_load_async_to_lds_b128 %0, %1, off offset:16"
                     :: "v"(lb), "v"(gb) : "memory");
    };

    v8f acc[4];
    const v8f zero8 = {0.f,0.f,0.f,0.f,0.f,0.f,0.f,0.f};
    #pragma unroll
    for (int j = 0; j < 4; ++j) acc[j] = zero8;

    stage(0, 0);
    asm volatile("s_wait_asynccnt 0x0" ::: "memory");
    __syncthreads();

    for (int k0 = 0; k0 < 512; k0 += 32) {
        const int buf = (k0 >> 5) & 1;
        if (k0 + 64 < 512) {                       // prefetch chunk after next into L2
            __builtin_prefetch(Ag + k0 + 64, 0, 3);
            __builtin_prefetch(Bg + k0 + 64, 0, 3);
        }
        if (k0 + 32 < 512) stage(buf ^ 1, k0 + 32);

        // ---- A fragment: two aligned 16B LDS loads (ds_load_b128)
        union { v16bf v; v8bf h[2]; } af;
        {
            const __bf16* Ar = &lA[buf][(wm * 16 + lm) * LDP];
            af.h[0] = *(const v8bf*)(Ar + kh * 8);
            af.h[1] = *(const v8bf*)(Ar + 16 + kh * 8);
        }
        // ---- load all 4 B fragments first, then 4 back-to-back WMMAs
        union { v16bf v; v8bf h[2]; } bfr[4];
        #pragma unroll
        for (int j = 0; j < 4; ++j) {
            const __bf16* Br = &lB[buf][(wn * 64 + j * 16 + lm) * LDP];
            bfr[j].h[0] = *(const v8bf*)(Br + kh * 8);
            bfr[j].h[1] = *(const v8bf*)(Br + 16 + kh * 8);
        }
        #pragma unroll
        for (int j = 0; j < 4; ++j) {
            acc[j] = __builtin_amdgcn_wmma_f32_16x16x32_bf16(
                         false, af.v, false, bfr[j].v, (short)0, acc[j], false, false);
        }

        asm volatile("s_wait_asynccnt 0x0" ::: "memory");
        __syncthreads();
    }

    // ---- epilogue: f32 C 16x16 layout (vgpr r, lane<16 -> M=r, lane>=16 -> M=8+r)
    #pragma unroll
    for (int j = 0; j < 4; ++j) {
        const int col = n0 + wn * 64 + j * 16 + lm;
        #pragma unroll
        for (int r = 0; r < 8; ++r) {
            const int row = m0 + wm * 16 + kh * 8 + r;
            C[(size_t)row * 2048 + col] = acc[j][r];
        }
    }
}

// ---------------------------------------------------------------- row LN over 2048 features (in place)
// out = scale * (d / sqrt(sum(d^2)+eps)) + shift   (sum, not mean, per reference)
__global__ __launch_bounds__(256) void ln_rows_2048_kernel(
    float* __restrict__ X, const float* __restrict__ scale, const float* __restrict__ shift)
{
    __shared__ float sm[256];
    float* xr = X + (size_t)blockIdx.x * 2048;
    const int tid = threadIdx.x;
    float v[8]; float s = 0.f;
    #pragma unroll
    for (int i = 0; i < 8; ++i) { v[i] = xr[tid + i * 256]; s += v[i]; }
    const float mean = block_reduce_sum(s, sm) * (1.0f / 2048.0f);
    float ss = 0.f;
    #pragma unroll
    for (int i = 0; i < 8; ++i) { v[i] -= mean; ss += v[i] * v[i]; }
    const float inv = 1.0f / sqrtf(block_reduce_sum(ss, sm) + 1e-5f);
    #pragma unroll
    for (int i = 0; i < 8; ++i) {
        const int c = tid + i * 256;
        xr[c] = scale[c] * (v[i] * inv) + shift[c];
    }
}

// ---------------------------------------------------------------- fused LSTM cell (one block per batch row)
__global__ __launch_bounds__(512) void lstm_cell_kernel(
    const float* __restrict__ gatesRaw,   // [64,2048] = h @ Wh^T (pre-LN)
    const float* __restrict__ gi_t,       // [64,2048] LN'ed input projection for step t
    const float* __restrict__ bias,
    const float* __restrict__ b2, const float* __restrict__ s2,
    const float* __restrict__ b3, const float* __restrict__ s3,
    float* __restrict__ cbuf, float* __restrict__ hbuf, __bf16* __restrict__ hbf,
    float* __restrict__ out, int t)
{
    __shared__ float sm[512];
    const int b = blockIdx.x;
    const int hcol = threadIdx.x;                 // 0..511
    const size_t r = (size_t)b * 2048;

    float g0 = gatesRaw[r + hcol];
    float g1 = gatesRaw[r + 512 + hcol];
    float g2 = gatesRaw[r + 1024 + hcol];
    float g3 = gatesRaw[r + 1536 + hcol];

    // LN over all 2048 gate pre-activations of this batch row
    const float mean = block_reduce_sum(g0 + g1 + g2 + g3, sm) * (1.0f / 2048.0f);
    g0 -= mean; g1 -= mean; g2 -= mean; g3 -= mean;
    const float inv = 1.0f / sqrtf(block_reduce_sum(g0*g0 + g1*g1 + g2*g2 + g3*g3, sm) + 1e-5f);

    const float xi = gi_t[r + hcol]        + b2[hcol]        * (g0 * inv) + s2[hcol]        + bias[hcol];
    const float xf = gi_t[r + 512 + hcol]  + b2[512 + hcol]  * (g1 * inv) + s2[512 + hcol]  + bias[512 + hcol];
    const float xg = gi_t[r + 1024 + hcol] + b2[1024 + hcol] * (g2 * inv) + s2[1024 + hcol] + bias[1024 + hcol];
    const float xo = gi_t[r + 1536 + hcol] + b2[1536 + hcol] * (g3 * inv) + s2[1536 + hcol] + bias[1536 + hcol];

    const float ig = sigmoidf_(xi);
    const float fg = sigmoidf_(xf);
    const float gg = tanhf(xg);
    const float og = sigmoidf_(xo);

    float z = fg * cbuf[b * 512 + hcol] + ig * gg;

    // LN over 512 cell features
    const float zm  = block_reduce_sum(z, sm) * (1.0f / 512.0f);
    const float zd  = z - zm;
    const float zin = 1.0f / sqrtf(block_reduce_sum(zd * zd, sm) + 1e-5f);
    const float cn  = b3[hcol] * (zd * zin) + s3[hcol];
    const float hn  = og * tanhf(cn);

    cbuf[b * 512 + hcol] = cn;
    hbuf[b * 512 + hcol] = hn;
    hbf[b * 512 + hcol]  = (__bf16)hn;            // bf16 copy feeds next step's WMMA GEMM
    out[(size_t)b * 262144 + (size_t)t * 512 + hcol] = hn;   // [B,T,H]
}

__global__ void write_final_kernel(const float* __restrict__ h, const float* __restrict__ c,
                                   float* __restrict__ tail) {
    int i = blockIdx.x * blockDim.x + threadIdx.x;
    if (i < 32768) { tail[i] = h[i]; tail[32768 + i] = c[i]; }
}

// ---------------------------------------------------------------- launcher
extern "C" void kernel_launch(void* const* d_in, const int* in_sizes, int n_in,
                              void* d_out, int out_size, void* d_ws, size_t ws_size,
                              hipStream_t stream) {
    const float* x    = (const float*)d_in[0];
    const float* hx   = (const float*)d_in[1];
    const float* cx   = (const float*)d_in[2];
    const float* Wi   = (const float*)d_in[3];
    const float* Wh   = (const float*)d_in[4];
    const float* bias = (const float*)d_in[5];
    const float* b1   = (const float*)d_in[6];
    const float* b2   = (const float*)d_in[7];
    const float* b3   = (const float*)d_in[8];
    const float* s1   = (const float*)d_in[9];
    const float* s2   = (const float*)d_in[10];
    const float* s3   = (const float*)d_in[11];
    float* out = (float*)d_out;

    char* ws = (char*)d_ws;
    float*  gi    = (float*)(ws);                           // 32768*2048 f32 = 256 MiB
    float*  gates = (float*)(ws + 268435456ull);            // 64*2048 f32
    float*  hbuf  = (float*)(ws + 268959744ull);            // 64*512 f32
    float*  cbuf  = (float*)(ws + 269090816ull);            // 64*512 f32
    __bf16* Wibf  = (__bf16*)(ws + 269221888ull);           // 2048*512 bf16 = 2 MiB
    __bf16* Whbf  = (__bf16*)(ws + 271319040ull);           // 2048*512 bf16 = 2 MiB
    __bf16* xbf   = (__bf16*)(ws + 273416192ull);           // 64*512*512 bf16 = 32 MiB
    __bf16* hbf   = (__bf16*)(ws + 306970624ull);           // 64*512 bf16 = 64 KiB

    const int nW = 2048 * 512;
    const int nX = 64 * 512 * 512;
    f32_to_bf16_kernel<<<(nW + 255) / 256, 256, 0, stream>>>(Wi, Wibf, nW);
    f32_to_bf16_kernel<<<(nW + 255) / 256, 256, 0, stream>>>(Wh, Whbf, nW);
    f32_to_bf16_kernel<<<(nX + 255) / 256, 256, 0, stream>>>(x, xbf, nX);
    init_hc_kernel<<<128, 256, 0, stream>>>(hx, cx, hbuf, hbf, cbuf, 32768);

    // gi = x @ Wi^T : M=32768 (r = t*64 + b -> x[b*T*I + t*I]), then LN(b1, s1)
    wmma_gemm_kernel<<<dim3(512, 16), 256, 0, stream>>>(xbf, 262144, 512, Wibf, gi);
    ln_rows_2048_kernel<<<32768, 256, 0, stream>>>(gi, b1, s1);

    // sequential scan
    for (int t = 0; t < 512; ++t) {
        wmma_gemm_kernel<<<dim3(1, 16), 256, 0, stream>>>(hbf, 512, 0, Whbf, gates);
        lstm_cell_kernel<<<64, 512, 0, stream>>>(gates, gi + (size_t)t * 131072,
                                                 bias, b2, s2, b3, s3,
                                                 cbuf, hbuf, hbf, out, t);
    }
    write_final_kernel<<<128, 256, 0, stream>>>(hbuf, cbuf, out + 16777216ull);
}
